// PointTransformerConv1_81698867904602
// MI455X (gfx1250) — matrix operand order repaired and checked by reading, not compiled, not run on hardware
//
#include <hip/hip_runtime.h>
#include <stdint.h>

typedef _Float16 v16h __attribute__((ext_vector_type(16)));
typedef _Float16 v8h  __attribute__((ext_vector_type(8)));
typedef float    v8f  __attribute__((ext_vector_type(8)));
typedef unsigned int u32x4 __attribute__((ext_vector_type(4)));
typedef int          i32x8 __attribute__((ext_vector_type(8)));
typedef int          i32x4 __attribute__((ext_vector_type(4)));

#define B_  8
#define N_  256
#define F_  59
#define C_  128
#define KP  64      // F padded to 64 for WMMA K
#define EPS_ 1e-5f

// ------------------------------------------------------------------
// TDM helpers: 1-D tensor DMA global -> LDS (CDNA5 Tensor Data Mover)
// D# layout per CDNA5 ISA:
//  group0: [1:0]=count=1, [63:32]=lds_addr, [120:64]=global_addr,
//          [127:126]=type=2
//  group1: [17:16]=data_size, [79:48]=tensor_dim0, [111:80]=tensor_dim1(=1),
//          [127:112]=tile_dim0, [143:128]=tile_dim1(0 => 1-D),
//          [207:160]=tensor_dim0_stride
// ------------------------------------------------------------------
static __device__ __forceinline__ uint32_t lds_off(const void* p) {
  // LDS aperture maps generic addr -> LDS offset via addr[31:0]
  return (uint32_t)(uintptr_t)p;
}

static __device__ __forceinline__ void tdm_load_1d(uint32_t ldsaddr,
                                                   const void* gptr,
                                                   uint32_t nelem,
                                                   uint32_t dszcode) {
  uint64_t ga = (uint64_t)(uintptr_t)gptr;
  u32x4 g0;
  g0[0] = 1u;                                        // count=1 (valid user D#)
  g0[1] = ldsaddr;                                   // lds_addr
  g0[2] = (uint32_t)ga;                              // global_addr[31:0]
  g0[3] = (uint32_t)((ga >> 32) & 0x01FFFFFFull)     // global_addr[56:32]
        | (2u << 30);                                // type=2 ("image")
  i32x8 g1;
  g1[0] = (int)(dszcode << 16);                      // wg_mask=0, data_size
  g1[1] = (int)((nelem & 0xFFFFu) << 16);            // tensor_dim0[15:0]
  g1[2] = (int)((nelem >> 16) | (1u << 16));         // tensor_dim0[31:16] | tensor_dim1=1
  g1[3] = (int)((nelem & 0xFFFFu) << 16);            // tile_dim0
  g1[4] = 0;                                         // tile_dim1=0 (1-D), tile_dim2=0
  g1[5] = (int)nelem;                                // tensor_dim0_stride[31:0]
  g1[6] = 0;
  g1[7] = 0;
  i32x4 gz = {0, 0, 0, 0};
#if __clang_major__ >= 23
  i32x8 gz8 = {0, 0, 0, 0, 0, 0, 0, 0};
  __builtin_amdgcn_tensor_load_to_lds(g0, g1, gz, gz, gz8, 0);
#else
  __builtin_amdgcn_tensor_load_to_lds(g0, g1, gz, gz, 0);
#endif
}

// ------------------------------------------------------------------
// Kernel 0: convert/pad fp32 -> f16 staging buffers in workspace
// ------------------------------------------------------------------
__global__ void cvt_kernel(const float* __restrict__ x,
                           const float* __restrict__ Wl,
                           const float* __restrict__ Ws,
                           const float* __restrict__ Wd,
                           const float* __restrict__ aw,
                           _Float16* __restrict__ xh,
                           _Float16* __restrict__ w3h,
                           _Float16* __restrict__ awh)
{
  const int NXH = B_*N_*KP;   // 131072
  const int NW  = 3*C_*KP;    // 24576
  const int NA  = C_*C_;      // 16384
  int idx = blockIdx.x*256 + threadIdx.x;
  if (idx < NXH) {
    int row = idx >> 6, k = idx & 63;
    xh[idx] = (k < F_) ? (_Float16)x[row*F_ + k] : (_Float16)0.f;
  } else if (idx < NXH+NW) {
    int t = idx - NXH;
    int which = t / (C_*KP);
    int rem = t % (C_*KP);
    int row = rem >> 6, k = rem & 63;
    const float* W = (which==0) ? Wl : ((which==1) ? Ws : Wd);
    w3h[t] = (k < F_) ? (_Float16)W[row*F_+k] : (_Float16)0.f;
  } else if (idx < NXH+NW+NA) {
    int t = idx - NXH - NW;
    awh[t] = (_Float16)aw[t];
  }
}

// ------------------------------------------------------------------
// Kernel 1: projections v / a_src / a_dst : [2048 x 64] x [64 x 128]
// ------------------------------------------------------------------
__global__ void __launch_bounds__(256) proj_kernel(
    const _Float16* __restrict__ xh, const _Float16* __restrict__ w3h,
    float* __restrict__ vO, float* __restrict__ asO, float* __restrict__ adO)
{
  const int m0   = blockIdx.x * 16;
  const int tid  = threadIdx.x;
  const int wave = tid >> 5;
  const int lane = tid & 31;
  const int hlf  = lane >> 4;
  const int lr   = lane & 15;

  v16h A[2];
  #pragma unroll
  for (int ks = 0; ks < 2; ++ks) {
    const _Float16* ap = xh + (m0 + lr)*KP + ks*32 + hlf*8;
    v8h lo = *(const v8h*)(ap);
    v8h hi = *(const v8h*)(ap + 16);
    A[ks] = __builtin_shufflevector(lo, hi, 0,1,2,3,4,5,6,7,8,9,10,11,12,13,14,15);
  }

  #pragma unroll
  for (int k = 0; k < 3; ++k) {
    int t    = wave*3 + k;     // 0..23
    int wsel = t >> 3;
    int nt   = t & 7;
    v8f d = {0.f,0.f,0.f,0.f,0.f,0.f,0.f,0.f};
    #pragma unroll
    for (int ks = 0; ks < 2; ++ks) {
      const _Float16* bp = w3h + (wsel*C_ + nt*16 + lr)*KP + ks*32 + hlf*8;
      v8h b0 = *(const v8h*)(bp);
      v8h b1 = *(const v8h*)(bp + 16);
      v16h Bf = __builtin_shufflevector(b0, b1, 0,1,2,3,4,5,6,7,8,9,10,11,12,13,14,15);
      d = __builtin_amdgcn_wmma_f32_16x16x32_f16(false, A[ks], false, Bf,
                                                 (short)0, d, false, false);
    }
    float* o = (wsel == 0) ? vO : ((wsel == 1) ? asO : adO);
    #pragma unroll
    for (int r = 0; r < 8; ++r) {
      int row = m0 + r + 8*hlf;
      o[row*C_ + nt*16 + lr] = d[r];
    }
  }
}

// ------------------------------------------------------------------
// Kernel 2: fused edge GEMM + per-channel online softmax + aggregate.
// One block per (batch b, 16-receiver tile). 8 waves; wave w owns
// receivers 2w, 2w+1. Bulk LDS staging via the Tensor Data Mover.
// Compute structure kept at the spill-free round-1 shape:
// per-receiver A fragments (32 VGPRs), B streamed from LDS.
// ------------------------------------------------------------------
__global__ void __launch_bounds__(256) attn_kernel(
    const float* __restrict__ pos,  const float* __restrict__ nrm,
    const float* __restrict__ posw_g, const float* __restrict__ posb_g,
    const float* __restrict__ pbg, const float* __restrict__ pbb,
    const float* __restrict__ pbm, const float* __restrict__ pbv,
    const float* __restrict__ attnb_g,
    const float* __restrict__ abg, const float* __restrict__ abb,
    const float* __restrict__ abm, const float* __restrict__ abv,
    const int*   __restrict__ rptr,
    const float* __restrict__ vG, const float* __restrict__ asG,
    const float* __restrict__ adG,
    const _Float16* __restrict__ awhG,
    float* __restrict__ out)
{
  __shared__ __align__(16) _Float16 awh[C_*C_];   // attn_w f16 [d][c]  32KB
  __shared__ float posw[C_*6];
  __shared__ float spos[C_], tpos[C_], sattn[C_], uattn[C_];
  __shared__ __align__(16) float adst[16*C_];
  __shared__ float Pis[16*C_], geoi[16*8];
  __shared__ __align__(16) float asrc[16*C_];
  __shared__ __align__(16) float vls[16*C_];
  __shared__ float Pjs[16*C_], geoj[16*8];

  const int b   = blockIdx.x >> 4;
  const int i0  = (blockIdx.x & 15) << 4;
  const int tid = threadIdx.x;
  const float rr = (float)rptr[0];
  const float r2 = rr*rr;

  // ---- prologue: TDM the big constant tiles, VALU the small ones ----
  if (tid < 32) {   // wave 0 issues the DMAs (wave-level, EXEC-ignored)
    tdm_load_1d(lds_off(awh),  awhG,                     C_*C_, 1); // f16
    tdm_load_1d(lds_off(adst), adG + ((b*N_ + i0) << 7), 16*C_, 2); // f32
  }
  #pragma unroll
  for (int q = 0; q < 3; ++q) posw[tid + 256*q] = posw_g[tid + 256*q];
  if (tid < C_) {
    float sp = pbg[tid] * rsqrtf(pbv[tid] + EPS_);
    spos[tid]  = sp;
    tpos[tid]  = pbb[tid] - pbm[tid]*sp;
    float sa = abg[tid] * rsqrtf(abv[tid] + EPS_);
    sattn[tid] = sa;
    uattn[tid] = attnb_g[tid]*sa + (abb[tid] - abm[tid]*sa);
  }
  if (tid < 16) {
    int g = b*N_ + i0 + tid;
    #pragma unroll
    for (int k = 0; k < 3; ++k) {
      geoi[tid*8+k]   = pos[g*3+k];
      geoi[tid*8+3+k] = nrm[g*3+k];
    }
  }
  if (tid < 32) __builtin_amdgcn_s_wait_tensorcnt(0);
  __syncthreads();

  // P_i[c] = <geo_i, pos_w[c]> + pos_b[c]
  #pragma unroll
  for (int q = 0; q < 8; ++q) {
    int idx = tid + 256*q;
    int il = idx >> 7, c = idx & 127;
    float s = posb_g[c];
    #pragma unroll
    for (int k = 0; k < 6; ++k) s += posw[c*6+k] * geoi[il*8+k];
    Pis[idx] = s;
  }

  const int wave = tid >> 5;
  const int lane = tid & 31;
  const int hlf  = lane >> 4;
  const int lr   = lane & 15;

  // online-softmax state: per (receiver 0/1, channel tile 0..7)
  float mS[16], lS[16], aS[16];
  #pragma unroll
  for (int s = 0; s < 16; ++s) { mS[s] = -__builtin_inff(); lS[s] = 0.f; aS[s] = 0.f; }

  for (int jt = 0; jt < 16; ++jt) {
    const int j0 = jt*16;
    __syncthreads();
    if (tid < 32) {   // TDM-stage a_src and v rows for this j-chunk
      tdm_load_1d(lds_off(asrc), asG + ((b*N_ + j0) << 7), 16*C_, 2);
      tdm_load_1d(lds_off(vls),  vG  + ((b*N_ + j0) << 7), 16*C_, 2);
    }
    if (tid < 16) {
      int g = b*N_ + j0 + tid;
      #pragma unroll
      for (int k = 0; k < 3; ++k) {
        geoj[tid*8+k]   = pos[g*3+k];
        geoj[tid*8+3+k] = nrm[g*3+k];
      }
    }
    if (tid < 32) __builtin_amdgcn_s_wait_tensorcnt(0);
    __syncthreads();
    #pragma unroll
    for (int q = 0; q < 8; ++q) {
      int idx = tid + 256*q;
      int jl = idx >> 7, c = idx & 127;
      float s = 0.f;
      #pragma unroll
      for (int k = 0; k < 6; ++k) s += posw[c*6+k] * geoj[jl*8+k];
      Pjs[idx] = s;
    }
    __syncthreads();

    #pragma unroll
    for (int ii = 0; ii < 2; ++ii) {
      const int il = wave*2 + ii;

      // radius mask distances for this lane's 8 D-rows (j = hlf*8 + r)
      float d2r[8];
      #pragma unroll
      for (int r = 0; r < 8; ++r) {
        int j = hlf*8 + r;
        float dx = geoi[il*8+0] - geoj[j*8+0];
        float dy = geoi[il*8+1] - geoj[j*8+1];
        float dz = geoi[il*8+2] - geoj[j*8+2];
        d2r[r] = dx*dx + dy*dy + dz*dz;
      }

      // A fragments: alpha_arg = a_dst_i - a_src_j + delta (rows j, cols c)
      v16h A[4];
      const int jr = lr;
      #pragma unroll
      for (int ks = 0; ks < 4; ++ks) {
        const int c0 = ks*32 + hlf*8;
        v16h af;
        #pragma unroll
        for (int u = 0; u < 8; ++u) {
          int c = c0 + u;
          float dlt = fmaxf((Pis[il*128+c] - Pjs[jr*128+c])*spos[c] + tpos[c], 0.f);
          af[u] = (_Float16)(adst[il*128+c] - asrc[jr*128+c] + dlt);
        }
        #pragma unroll
        for (int u = 0; u < 8; ++u) {
          int c = c0 + 16 + u;
          float dlt = fmaxf((Pis[il*128+c] - Pjs[jr*128+c])*spos[c] + tpos[c], 0.f);
          af[8+u] = (_Float16)(adst[il*128+c] - asrc[jr*128+c] + dlt);
        }
        A[ks] = af;
      }

      #pragma unroll
      for (int nt = 0; nt < 8; ++nt) {
        v8f d = {0.f,0.f,0.f,0.f,0.f,0.f,0.f,0.f};
        #pragma unroll
        for (int ks = 0; ks < 4; ++ks) {
          const _Float16* bp = &awh[(nt*16 + lr)*C_ + ks*32 + hlf*8];
          v8h b0 = *(const v8h*)(bp);
          v8h b1 = *(const v8h*)(bp + 16);
          v16h Bf = __builtin_shufflevector(b0, b1, 0,1,2,3,4,5,6,7,8,9,10,11,12,13,14,15);
          d = __builtin_amdgcn_wmma_f32_16x16x32_f16(false, A[ks], false, Bf,
                                                     (short)0, d, false, false);
        }
        // per-channel online softmax; lane pair (lr, lr+16) share channel c
        const int c   = nt*16 + lr;
        const float sa = sattn[c], ua = uattn[c];
        const float sp = spos[c],  tp = tpos[c];
        const float pic = Pis[il*128+c];
        float sc[8];
        float cmax = -__builtin_inff();
        #pragma unroll
        for (int r = 0; r < 8; ++r) {
          float s = fmaxf(d[r]*sa + ua, 0.f);
          if (d2r[r] > r2) s = -__builtin_inff();
          sc[r] = s;
          cmax = fmaxf(cmax, s);
        }
        cmax = fmaxf(cmax, __shfl_xor(cmax, 16, 32));
        const int st = ii*8 + nt;
        float nm = fmaxf(mS[st], cmax);
        if (nm != -__builtin_inff()) {
          float scl = __expf(mS[st] - nm);
          lS[st] *= scl; aS[st] *= scl; mS[st] = nm;
          #pragma unroll
          for (int r = 0; r < 8; ++r) {
            if (sc[r] != -__builtin_inff()) {
              float p = __expf(sc[r] - nm);
              int j = hlf*8 + r;
              float dlt = fmaxf((pic - Pjs[j*128+c])*sp + tp, 0.f);
              aS[st] += p * (vls[j*128+c] + dlt);
              lS[st] += p;
            }
          }
        }
      }
    }
  }

  // ---- epilogue: merge lane partners and store ----
  #pragma unroll
  for (int ii = 0; ii < 2; ++ii) {
    int il = wave*2 + ii;
    #pragma unroll
    for (int nt = 0; nt < 8; ++nt) {
      int st = ii*8 + nt;
      float lt = lS[st] + __shfl_xor(lS[st], 16, 32);
      float at = aS[st] + __shfl_xor(aS[st], 16, 32);
      if (hlf == 0) {
        out[((b*N_ + i0 + il) << 7) + nt*16 + lr] = at / lt;
      }
    }
  }
}

// ------------------------------------------------------------------
extern "C" void kernel_launch(void* const* d_in, const int* in_sizes, int n_in,
                              void* d_out, int out_size, void* d_ws, size_t ws_size,
                              hipStream_t stream) {
  const float* x    = (const float*)d_in[0];
  const float* pos  = (const float*)d_in[1];
  const float* nrm  = (const float*)d_in[2];
  const float* Wl   = (const float*)d_in[3];
  const float* Ws   = (const float*)d_in[4];
  const float* Wd   = (const float*)d_in[5];
  const float* posw = (const float*)d_in[6];
  const float* posb = (const float*)d_in[7];
  const float* pbg  = (const float*)d_in[8];
  const float* pbb  = (const float*)d_in[9];
  const float* pbm  = (const float*)d_in[10];
  const float* pbv  = (const float*)d_in[11];
  const float* aw   = (const float*)d_in[12];
  const float* ab   = (const float*)d_in[13];
  const float* abg  = (const float*)d_in[14];
  const float* abb  = (const float*)d_in[15];
  const float* abm  = (const float*)d_in[16];
  const float* abv  = (const float*)d_in[17];
  const int*   rp   = (const int*)  d_in[18];
  float* out = (float*)d_out;

  uint8_t* w = (uint8_t*)d_ws;
  float*    vW  = (float*)(w);                           // 1 MiB
  float*    asW = (float*)(w + (1u<<20));                // 1 MiB
  float*    adW = (float*)(w + (2u<<20));                // 1 MiB
  _Float16* xh  = (_Float16*)(w + (3u<<20));             // 256 KiB
  _Float16* w3h = (_Float16*)(w + (3u<<20) + 262144u);   // 48 KiB
  _Float16* awh = (_Float16*)(w + (3u<<20) + 262144u + 49152u); // 32 KiB

  cvt_kernel<<<672, 256, 0, stream>>>(x, Wl, Ws, Wd, aw, xh, w3h, awh);
  proj_kernel<<<(B_*N_)/16, 256, 0, stream>>>(xh, w3h, vW, asW, adW);
  attn_kernel<<<B_*(N_/16), 256, 0, stream>>>(pos, nrm, posw, posb,
                                              pbg, pbb, pbm, pbv,
                                              ab, abg, abb, abm, abv,
                                              rp, vW, asW, adW, awh, out);
}
